// MixtralHeadless_5523327943200
// MI455X (gfx1250) — matrix operand order repaired and checked by reading, not compiled, not run on hardware
//
#include <hip/hip_runtime.h>
#include <hip/hip_bf16.h>
#include <math.h>

// ---------------------------------------------------------------------------
// Mixtral-headless forward on gfx1250 (MI455X).
//  - GEMMs: v_wmma_f32_16x16x32_bf16, wave32. Block = 4 waves, 64x128 C tile;
//    each wave does a 4x2 WMMA tile grid (8 wmma / 32-K step).
//  - A tile staged to LDS via async global->LDS (ASYNCcnt path), shared by
//    all 4 waves; bank-conflict-free padded layout for ds_load_b128 frags.
//  - B operands pre-converted to bf16, K-major, so fragment loads are b128.
//  - Residual adds + MoE routing weights fused in the GEMM epilogue.
// ---------------------------------------------------------------------------

#define SEQ 2048
#define DIM 1024
#define NH 16
#define NKV 8
#define HD 64
#define NL 2
#define FFD 3584
#define NE 8

#define KSTEP 64
#define LROW 72            // padded LDS row stride (elements): 36 banks/row

typedef __attribute__((ext_vector_type(16))) __bf16 v16bf;
typedef __attribute__((ext_vector_type(16))) unsigned short ushort16;
typedef __attribute__((ext_vector_type(8)))  unsigned short ushort8;
typedef __attribute__((ext_vector_type(8)))  float v8f;

#if defined(__has_builtin)
#  if __has_builtin(__builtin_amdgcn_global_load_async_to_lds_b128)
#    define HAVE_ASYNC_LDS 1
#  else
#    define HAVE_ASYNC_LDS 0
#  endif
#else
#  define HAVE_ASYNC_LDS 0
#endif

#if HAVE_ASYNC_LDS
typedef __attribute__((ext_vector_type(4))) int v4i;
typedef __attribute__((address_space(1))) v4i* as1_v4i_p;   // global v4i*
typedef __attribute__((address_space(3))) v4i* as3_v4i_p;   // LDS v4i*
#  if __has_builtin(__builtin_amdgcn_s_wait_asynccnt)
#    define WAIT_ASYNC() __builtin_amdgcn_s_wait_asynccnt(0)
#  else
#    define WAIT_ASYNC() asm volatile("s_wait_asynccnt 0" ::: "memory")
#  endif
#endif

__device__ inline unsigned short f32_to_bf16(float f) {
  unsigned int u = __builtin_bit_cast(unsigned int, f);
  unsigned int r = u + 0x7FFFu + ((u >> 16) & 1u);   // round-to-nearest-even
  return (unsigned short)(r >> 16);
}

// -------------------------- embedding gather -------------------------------
__global__ void embed_kernel(const int* __restrict__ x,
                             const float* __restrict__ emb,
                             float* __restrict__ h) {
  int s = blockIdx.x;
  int tok = x[s];
  const float* src = emb + (size_t)tok * DIM;
  float* dst = h + (size_t)s * DIM;
  for (int d = threadIdx.x; d < DIM; d += blockDim.x) dst[d] = src[d];
}

// ------------------------------ RMSNorm ------------------------------------
__global__ void rmsnorm_kernel(const float* __restrict__ x,
                               const float* __restrict__ w,
                               float* __restrict__ out) {
  int t = blockIdx.x;
  const float* xp = x + (size_t)t * DIM;
  float* op = out + (size_t)t * DIM;
  __shared__ float red[8];
  float ss = 0.f;
  for (int d = threadIdx.x; d < DIM; d += blockDim.x) {
    float v = xp[d];
    ss += v * v;
  }
#pragma unroll
  for (int off = 16; off >= 1; off >>= 1) ss += __shfl_xor(ss, off, 32);
  int wv = threadIdx.x >> 5, ln = threadIdx.x & 31;
  if (ln == 0) red[wv] = ss;
  __syncthreads();
  if (threadIdx.x == 0) {
    float tot = 0.f;
#pragma unroll
    for (int i = 0; i < 8; ++i) tot += red[i];
    red[0] = tot;
  }
  __syncthreads();
  float inv = rsqrtf(red[0] / (float)DIM + 1e-5f);
  for (int d = threadIdx.x; d < DIM; d += blockDim.x)
    op[d] = xp[d] * inv * w[d];
}

// -------------------- elementwise f32 -> bf16 convert ----------------------
__global__ void convert_bf16_kernel(const float* __restrict__ in,
                                    unsigned short* __restrict__ out, size_t n) {
  size_t i = (size_t)blockIdx.x * blockDim.x + threadIdx.x;
  size_t stride = (size_t)gridDim.x * blockDim.x;
  for (; i < n; i += stride) out[i] = f32_to_bf16(in[i]);
}

// ---------------- LDS-tiled transpose + convert (KxN -> NxK) ---------------
__global__ void transpose_convert_kernel(const float* __restrict__ in,  // Kd x N
                                         unsigned short* __restrict__ out, // N x Kd
                                         int Kd, int N) {
  __shared__ float tile[32][33];
  int kb = blockIdx.y * 32, nb = blockIdx.x * 32;
  int tx = threadIdx.x & 31, ty = threadIdx.x >> 5;   // 32 x 8
#pragma unroll
  for (int i = ty; i < 32; i += 8)
    tile[i][tx] = in[(size_t)(kb + i) * N + nb + tx];
  __syncthreads();
#pragma unroll
  for (int i = ty; i < 32; i += 8)
    out[(size_t)(nb + i) * Kd + kb + tx] = f32_to_bf16(tile[tx][i]);
}

// ------------------------- WMMA bf16 GEMM ----------------------------------
// C[M,N] (op)= rowScale[m] * (A x B^T):  A bf16 [M,Kd], B bf16 [N,Kd].
// Block: 4 waves, 64(M) x 128(N) C tile. A tile staged in LDS (shared).
// Requires M%64==0, N%128==0, Kd%64==0.
__device__ inline v16bf frag_from_pair(const unsigned short* p) { // K..K+7, K+16..K+23
  ushort8 lo = *(const ushort8*)(p);
  ushort8 hi = *(const ushort8*)(p + 16);
  ushort16 t = __builtin_shufflevector(lo, hi, 0, 1, 2, 3, 4, 5, 6, 7,
                                       8, 9, 10, 11, 12, 13, 14, 15);
  return __builtin_bit_cast(v16bf, t);
}
__device__ inline v16bf frag_contig(const unsigned short* p) {   // K..K+15
  ushort16 t = *(const ushort16*)(p);
  return __builtin_bit_cast(v16bf, t);
}

template <bool HS, bool ACC>
__global__ void gemm_bf16(const unsigned short* __restrict__ A,
                          const unsigned short* __restrict__ B,
                          float* __restrict__ C,
                          int M, int N, int Kd,
                          const float* __restrict__ rowScale, int rsStride) {
  __shared__ unsigned short tileA[64 * LROW];   // 64 rows x 64 K (padded)

  int lane = threadIdx.x & 31;
  int wave = threadIdx.x >> 5;
  int nbase = (blockIdx.x * 4 + wave) * 32;
  int mbase = blockIdx.y * 64;

  int akb = (lane >> 4) * 8;      // A frag: lanes 16-31 take K offset +8
  int bkb = (lane >> 4) * 8;      // B frag: lanes 16-31 take K offset +8

  const unsigned short* b0p = B + (size_t)(nbase + (lane & 15)) * Kd;
  const unsigned short* b1p = b0p + (size_t)16 * Kd;

  // staging map: chunk c = tid*4+j ; row = c>>3, K-chunk = (c&7)*8
  int c0 = threadIdx.x * 4;
  int srow[4], skc[4];
#pragma unroll
  for (int j = 0; j < 4; ++j) {
    srow[j] = (c0 + j) >> 3;
    skc[j]  = ((c0 + j) & 7) * 8;
  }

  v8f acc[4][2] = {};

  for (int k0 = 0; k0 < Kd; k0 += KSTEP) {
    // ---- stage A tile (64 x 64) cooperatively ----
#pragma unroll
    for (int j = 0; j < 4; ++j) {
      const unsigned short* g = A + (size_t)(mbase + srow[j]) * Kd + k0 + skc[j];
      unsigned short* l = &tileA[srow[j] * LROW + skc[j]];
#if HAVE_ASYNC_LDS
      __builtin_amdgcn_global_load_async_to_lds_b128(
          (as1_v4i_p)(unsigned short*)(g), (as3_v4i_p)(l), 0, 0);
#else
      *(ushort8*)l = *(const ushort8*)g;
#endif
    }
#if HAVE_ASYNC_LDS
    WAIT_ASYNC();
#endif
    __syncthreads();

    // prefetch B two K-steps ahead
    __builtin_prefetch(b0p + k0 + 2 * KSTEP, 0, 1);
    __builtin_prefetch(b1p + k0 + 2 * KSTEP, 0, 1);

#pragma unroll
    for (int kk = 0; kk < KSTEP; kk += 32) {
      v16bf b0 = frag_from_pair(b0p + k0 + kk + bkb);
      v16bf b1 = frag_from_pair(b1p + k0 + kk + bkb);
#pragma unroll
      for (int mt = 0; mt < 4; ++mt) {
        const unsigned short* ap =
            &tileA[(mt * 16 + (lane & 15)) * LROW + kk + akb];
        v16bf a = frag_from_pair(ap);     // ds_load_b128 x2, conflict-free
        acc[mt][0] = __builtin_amdgcn_wmma_f32_16x16x32_bf16(
            false, a, false, b0, (short)0, acc[mt][0], false, false);
        acc[mt][1] = __builtin_amdgcn_wmma_f32_16x16x32_bf16(
            false, a, false, b1, (short)0, acc[mt][1], false, false);
      }
    }
    __syncthreads();
  }

  // C/D layout: VGPR r, lane l -> row = (l>=16)*8 + r, col = l&15
  int colq = lane & 15;
  int rsub = (lane >> 4) * 8;
#pragma unroll
  for (int mt = 0; mt < 4; ++mt) {
    int rbase = mbase + mt * 16 + rsub;
    float sc[8];
    if (HS) {
#pragma unroll
      for (int r = 0; r < 8; ++r) sc[r] = rowScale[(size_t)(rbase + r) * rsStride];
    }
#pragma unroll
    for (int nt = 0; nt < 2; ++nt) {
      int col = nbase + nt * 16 + colq;
#pragma unroll
      for (int r = 0; r < 8; ++r) {
        float v = acc[mt][nt][r];
        if (HS) v *= sc[r];
        size_t idx = (size_t)(rbase + r) * N + col;
        if (ACC) v += C[idx];
        C[idx] = v;
      }
    }
  }
}

// ------------------------------- RoPE --------------------------------------
__global__ void rope_kernel(float* __restrict__ x, int nheads) {
  int s = blockIdx.x, h = blockIdx.y, i = threadIdx.x;   // i in 0..31
  float freq = powf(1.0e6f, -2.0f * (float)i / (float)HD);
  float ang = (float)s * freq;
  float cs = cosf(ang), sn = sinf(ang);
  float* p = x + ((size_t)s * nheads + h) * HD;
  float x1 = p[2 * i], x2 = p[2 * i + 1];
  p[2 * i]     = x1 * cs - x2 * sn;
  p[2 * i + 1] = x1 * sn + x2 * cs;
}

// --------------------- causal attention (streaming softmax) ----------------
__global__ void attention_kernel(const float* __restrict__ q,
                                 const float* __restrict__ k,
                                 const float* __restrict__ v,
                                 float* __restrict__ o) {
  int qi = blockIdx.x;
  int hh = blockIdx.y;
  int lane = threadIdx.x;
  int kvh = hh >> 1;
  const float scale = 0.125f;

  const float* qp = q + ((size_t)qi * NH + hh) * HD;
  float q0 = qp[2 * lane], q1 = qp[2 * lane + 1];

  float m = -1e30f, l = 0.f, a0 = 0.f, a1 = 0.f;
  for (int ki = 0; ki <= qi; ++ki) {
    const float* kp = k + ((size_t)ki * NKV + kvh) * HD;
    float p = q0 * kp[2 * lane] + q1 * kp[2 * lane + 1];
#pragma unroll
    for (int off = 16; off >= 1; off >>= 1) p += __shfl_xor(p, off, 32);
    float s = p * scale;
    float nm = fmaxf(m, s);
    float f = expf(m - nm);
    float w = expf(s - nm);
    const float* vp = v + ((size_t)ki * NKV + kvh) * HD;
    a0 = a0 * f + w * vp[2 * lane];
    a1 = a1 * f + w * vp[2 * lane + 1];
    l = l * f + w;
    m = nm;
  }
  float* op = o + ((size_t)qi * NH + hh) * HD;
  float inv = 1.0f / l;
  op[2 * lane]     = a0 * inv;
  op[2 * lane + 1] = a1 * inv;
}

// ------------------------ MoE gate: top-2 weights --------------------------
__global__ void gate_kernel(const float* __restrict__ xn,
                            const float* __restrict__ gw,
                            float* __restrict__ cw) {
  int t = blockIdx.x;
  int lane = threadIdx.x;
  const float* xp = xn + (size_t)t * DIM;
  float acc[NE];
#pragma unroll
  for (int e = 0; e < NE; ++e) acc[e] = 0.f;
  for (int d = lane; d < DIM; d += 32) {
    float xv = xp[d];
#pragma unroll
    for (int e = 0; e < NE; ++e) acc[e] += xv * gw[(size_t)e * DIM + d];
  }
#pragma unroll
  for (int e = 0; e < NE; ++e) {
#pragma unroll
    for (int off = 16; off >= 1; off >>= 1) acc[e] += __shfl_xor(acc[e], off, 32);
  }
  if (lane == 0) {
    int i1 = 0;
#pragma unroll
    for (int e = 1; e < NE; ++e) if (acc[e] > acc[i1]) i1 = e;
    int i2 = (i1 == 0) ? 1 : 0;
#pragma unroll
    for (int e = 0; e < NE; ++e)
      if (e != i1 && acc[e] > acc[i2]) i2 = e;
    float p1 = 1.0f / (1.0f + expf(acc[i2] - acc[i1]));
    float* cwp = cw + (size_t)t * NE;
#pragma unroll
    for (int e = 0; e < NE; ++e) cwp[e] = 0.f;
    cwp[i1] = p1;
    cwp[i2] = 1.0f - p1;
  }
}

// -------------------------- silu(h1) * h3 ----------------------------------
__global__ void silu_mul_kernel(float* __restrict__ h1,
                                const float* __restrict__ h3, size_t n) {
  size_t i = (size_t)blockIdx.x * blockDim.x + threadIdx.x;
  size_t stride = (size_t)gridDim.x * blockDim.x;
  for (; i < n; i += stride) {
    float a = h1[i];
    float s = a / (1.0f + expf(-a));
    h1[i] = s * h3[i];
  }
}

// --------------------------- host orchestration ----------------------------
template <bool HS, bool ACC>
static inline void launch_gemm(const unsigned short* A, const unsigned short* B,
                               float* C, int M, int N, int K,
                               const float* rowScale, int rsStride, hipStream_t st) {
  dim3 grid(N / 128, M / 64);
  gemm_bf16<HS, ACC><<<grid, 128, 0, st>>>(A, B, C, M, N, K, rowScale, rsStride);
}

static inline void launch_convert(const float* in, unsigned short* out, size_t n,
                                  hipStream_t st) {
  convert_bf16_kernel<<<2048, 256, 0, st>>>(in, out, n);
}

static inline void launch_transpose(const float* in, unsigned short* out,
                                    int Kd, int N, hipStream_t st) {
  transpose_convert_kernel<<<dim3(N / 32, Kd / 32), 256, 0, st>>>(in, out, Kd, N);
}

extern "C" void kernel_launch(void* const* d_in, const int* in_sizes, int n_in,
                              void* d_out, int out_size, void* d_ws, size_t ws_size,
                              hipStream_t stream) {
  const int*   x       = (const int*)d_in[0];
  const float* emb     = (const float*)d_in[1];
  const float* ln_w    = (const float*)d_in[2];
  const float* ffln_w  = (const float*)d_in[3];
  const float* wq      = (const float*)d_in[4];
  const float* wk      = (const float*)d_in[5];
  const float* wv      = (const float*)d_in[6];
  const float* wo      = (const float*)d_in[7];
  const float* gate_w  = (const float*)d_in[8];
  const float* w1      = (const float*)d_in[9];
  const float* w2      = (const float*)d_in[10];
  const float* w3      = (const float*)d_in[11];
  const float* dnorm_w = (const float*)d_in[12];
  float* out = (float*)d_out;

  // fp32 scratch
  float* f = (float*)d_ws;
  float* h   = f;  f += (size_t)SEQ * DIM;
  float* xn  = f;  f += (size_t)SEQ * DIM;
  float* q   = f;  f += (size_t)SEQ * NH  * HD;
  float* kb  = f;  f += (size_t)SEQ * NKV * HD;
  float* vb  = f;  f += (size_t)SEQ * NKV * HD;
  float* ob  = f;  f += (size_t)SEQ * NH  * HD;
  float* h1  = f;  f += (size_t)SEQ * FFD;
  float* h3  = f;  f += (size_t)SEQ * FFD;
  float* cw  = f;  f += (size_t)SEQ * NE;
  // bf16 scratch
  unsigned short* u = (unsigned short*)f;
  unsigned short* xn_bf = u;  u += (size_t)SEQ * DIM;
  unsigned short* ob_bf = u;  u += (size_t)SEQ * NH * HD;
  unsigned short* h1_bf = u;  u += (size_t)SEQ * FFD;
  unsigned short* wbuf  = u;  u += (size_t)FFD * DIM;   // shared weight staging

  embed_kernel<<<SEQ, 256, 0, stream>>>(x, emb, h);

  for (int i = 0; i < NL; ++i) {
    const float* wq_i = wq + (size_t)i * DIM * (NH * HD);
    const float* wk_i = wk + (size_t)i * DIM * (NKV * HD);
    const float* wv_i = wv + (size_t)i * DIM * (NKV * HD);
    const float* wo_i = wo + (size_t)i * (NH * HD) * DIM;

    // ---- attention block ----
    rmsnorm_kernel<<<SEQ, 256, 0, stream>>>(h, ln_w + (size_t)i * DIM, xn);
    launch_convert(xn, xn_bf, (size_t)SEQ * DIM, stream);

    launch_transpose(wq_i, wbuf, DIM, NH * HD, stream);
    launch_gemm<false, false>(xn_bf, wbuf, q, SEQ, NH * HD, DIM, nullptr, 0, stream);
    launch_transpose(wk_i, wbuf, DIM, NKV * HD, stream);
    launch_gemm<false, false>(xn_bf, wbuf, kb, SEQ, NKV * HD, DIM, nullptr, 0, stream);
    launch_transpose(wv_i, wbuf, DIM, NKV * HD, stream);
    launch_gemm<false, false>(xn_bf, wbuf, vb, SEQ, NKV * HD, DIM, nullptr, 0, stream);

    rope_kernel<<<dim3(SEQ, NH),  32, 0, stream>>>(q,  NH);
    rope_kernel<<<dim3(SEQ, NKV), 32, 0, stream>>>(kb, NKV);
    attention_kernel<<<dim3(SEQ, NH), 32, 0, stream>>>(q, kb, vb, ob);

    // h += ob @ wo  (fused residual add)
    launch_convert(ob, ob_bf, (size_t)SEQ * NH * HD, stream);
    launch_transpose(wo_i, wbuf, NH * HD, DIM, stream);
    launch_gemm<false, true>(ob_bf, wbuf, h, SEQ, DIM, NH * HD, nullptr, 0, stream);

    // ---- MoE block (dense over experts, scaled by routing weight) ----
    rmsnorm_kernel<<<SEQ, 256, 0, stream>>>(h, ffln_w + (size_t)i * DIM, xn);
    launch_convert(xn, xn_bf, (size_t)SEQ * DIM, stream);
    gate_kernel<<<SEQ, 32, 0, stream>>>(xn, gate_w + (size_t)i * NE * DIM, cw);

    for (int e = 0; e < NE; ++e) {
      const float* w1_e = w1 + ((size_t)i * NE + e) * (size_t)FFD * DIM;
      const float* w3_e = w3 + ((size_t)i * NE + e) * (size_t)FFD * DIM;
      const float* w2_e = w2 + ((size_t)i * NE + e) * (size_t)DIM * FFD;

      launch_convert(w1_e, wbuf, (size_t)FFD * DIM, stream);      // already NxK
      launch_gemm<false, false>(xn_bf, wbuf, h1, SEQ, FFD, DIM, nullptr, 0, stream);
      launch_convert(w3_e, wbuf, (size_t)FFD * DIM, stream);
      launch_gemm<false, false>(xn_bf, wbuf, h3, SEQ, FFD, DIM, nullptr, 0, stream);

      silu_mul_kernel<<<1024, 256, 0, stream>>>(h1, h3, (size_t)SEQ * FFD);
      launch_convert(h1, h1_bf, (size_t)SEQ * FFD, stream);

      // h += cw[:,e] * (h1 @ w2^T)   (fused residual + routing weight)
      launch_convert(w2_e, wbuf, (size_t)DIM * FFD, stream);      // already NxK
      launch_gemm<true, true>(h1_bf, wbuf, h, SEQ, DIM, FFD, cw + e, NE, stream);
    }
  }

  rmsnorm_kernel<<<SEQ, 256, 0, stream>>>(h, dnorm_w, out);
}
